// TrellisLinear_62989990363306
// MI455X (gfx1250) — matrix operand order repaired and checked by reading, not compiled, not run on hardware
//
#include <hip/hip_runtime.h>
#include <hip/hip_bf16.h>

typedef __attribute__((ext_vector_type(16))) _Float16 v16h;
typedef __attribute__((ext_vector_type(4)))  _Float16 v4h;
typedef __attribute__((ext_vector_type(8)))  float    v8f;
typedef __attribute__((ext_vector_type(8)))  unsigned int v8u;

#define IN_DIM  4096
#define OUT_DIM 4096
#define MROWS   8192
#define NSEQ    32768
#define TSTEPS  256

// ---------------------------------------------------------------------------
// Kernel 1: trellis decode.  Lookup masks state to 10 bits -> state after
// step t depends only on the last 5 two-bit codes; every (seq, t) independent.
// ---------------------------------------------------------------------------
__global__ __launch_bounds__(256) void decode_kernel(
    const int* __restrict__ trellis,      // [NSEQ][16]
    const float* __restrict__ tlut,       // [1024][2]
    float* __restrict__ w_hat)            // [OUT][IN] flat
{
    const int s = blockIdx.x;             // sequence
    const int t = threadIdx.x;            // step
    const int* words = trellis + s * 16;

    int state = 0;
    int j0 = t - 4; if (j0 < 0) j0 = 0;
    #pragma unroll 5
    for (int j = j0; j <= t; ++j) {
        int code = (words[j >> 4] >> (30 - 2 * (j & 15))) & 3;
        state = ((state << 2) | code) & 1023;
    }
    float2 v = ((const float2*)tlut)[state];
    ((float2*)w_hat)[(size_t)s * TSTEPS + t] = v;
}

// ---------------------------------------------------------------------------
// Kernel 2: fp32 -> f16 convert for x (4 elems / thread).
// ---------------------------------------------------------------------------
__global__ __launch_bounds__(256) void cvt_kernel(
    const float* __restrict__ in, _Float16* __restrict__ out)
{
    size_t i = ((size_t)blockIdx.x * 256 + threadIdx.x) * 4;
    float4 v = *(const float4*)(in + i);
    v4h h; h.x = (_Float16)v.x; h.y = (_Float16)v.y;
    h.z = (_Float16)v.z; h.w = (_Float16)v.w;
    *(v4h*)(out + i) = h;
}

// ---------------------------------------------------------------------------
// Kernel 3: FWHT along rows (IN dim), then scale column j by SU[j] (commutes
// with the later column transform). One block per row, 16KB LDS.
// ---------------------------------------------------------------------------
__global__ __launch_bounds__(256) void fwht_row_kernel(
    const float* __restrict__ in, float* __restrict__ out,
    const float* __restrict__ colScale)
{
    __shared__ float s[IN_DIM];
    const int row = blockIdx.x;
    const int tid = threadIdx.x;

    for (int i = tid; i < IN_DIM; i += 256)
        s[i] = in[(size_t)row * IN_DIM + i];
    __syncthreads();

    for (int h = 1; h < IN_DIM; h <<= 1) {
        #pragma unroll
        for (int p = tid; p < IN_DIM / 2; p += 256) {
            int idx = ((p & ~(h - 1)) << 1) | (p & (h - 1));
            float a = s[idx], b = s[idx + h];
            s[idx] = a + b; s[idx + h] = a - b;
        }
        __syncthreads();
    }
    const float inv = 1.0f / 64.0f;        // 1/sqrt(4096)
    for (int i = tid; i < IN_DIM; i += 256)
        out[(size_t)row * IN_DIM + i] = s[i] * inv * colScale[i];
}

// ---------------------------------------------------------------------------
// Kernel 4: FWHT along columns (OUT dim), scale row n by SV[n], store f16.
// ---------------------------------------------------------------------------
__global__ __launch_bounds__(256) void fwht_col_kernel(
    const float* __restrict__ in, _Float16* __restrict__ out,
    const float* __restrict__ rowScale)
{
    __shared__ float s[OUT_DIM];
    const int col = blockIdx.x;
    const int tid = threadIdx.x;

    for (int i = tid; i < OUT_DIM; i += 256)
        s[i] = in[(size_t)i * IN_DIM + col];
    __syncthreads();

    for (int h = 1; h < OUT_DIM; h <<= 1) {
        #pragma unroll
        for (int p = tid; p < OUT_DIM / 2; p += 256) {
            int idx = ((p & ~(h - 1)) << 1) | (p & (h - 1));
            float a = s[idx], b = s[idx + h];
            s[idx] = a + b; s[idx + h] = a - b;
        }
        __syncthreads();
    }
    const float inv = 1.0f / 64.0f;
    for (int i = tid; i < OUT_DIM; i += 256)
        out[(size_t)i * IN_DIM + col] = (_Float16)(s[i] * inv * rowScale[i]);
}

// ---------------------------------------------------------------------------
// Kernel 5: WMMA GEMM  y[m,n] = sum_k xh[m,k] * Wh[n,k]   (fp32 accum)
// Block tile 128x128, 8 waves (wave tile 64x32), K-tile 32.
// 3-stage software pipeline: LDS filled by GLOBAL_LOAD_ASYNC_TO_LDS_B128;
// each iteration issues tile kt+2 and waits ASYNCcnt<=4 (async loads retire
// in order; 4 per wave per tile), so every tile gets a full iteration of
// WMMA work to hide its fill latency.
// ---------------------------------------------------------------------------
#define GM 128
#define GN 128
#define GK 32
#define LDT 40   // GK + 8 halves padding, keeps 16B alignment
#define ABUF (GM * LDT)   // halves per A stage buffer

// LDS byte offset of a __shared__ pointer: generic LDS pointers are
// {aperture_hi32, offset32}, so the low 32 bits are the LDS offset.
__device__ __forceinline__ unsigned lds_off(const void* p) {
    return (unsigned)(unsigned long long)p;
}

__device__ __forceinline__ void async_ld_b128(unsigned lds_byte_off,
                                              const void* gptr) {
    asm volatile("global_load_async_to_lds_b128 %0, %1, off"
                 :: "v"(lds_byte_off),
                    "v"((unsigned long long)gptr)
                 : "memory");
}

__device__ __forceinline__ void wait_asynccnt(int which) {
    if (which == 0) asm volatile("s_wait_asynccnt 0x0" ::: "memory");
    else            asm volatile("s_wait_asynccnt 0x4" ::: "memory");
}

__global__ __launch_bounds__(256) void gemm_kernel(
    const _Float16* __restrict__ A,   // [MROWS][K]
    const _Float16* __restrict__ B,   // [N][K]  (== W row-major)
    float* __restrict__ C)            // [MROWS][N]
{
    constexpr int K = IN_DIM, N = OUT_DIM;
    __shared__ alignas(16) _Float16 Ash[3][ABUF];
    __shared__ alignas(16) _Float16 Bsh[3][ABUF];

    const int tid  = threadIdx.x;
    const int lane = tid & 31;
    const int wave = tid >> 5;
    const int wm   = (wave >> 2) * 64;    // wave M offset within block tile
    const int wn   = (wave & 3) * 32;     // wave N offset within block tile
    const int m0   = blockIdx.x * GM;
    const int n0   = blockIdx.y * GN;
    const int l15  = lane & 15;
    const int lhi  = lane >> 4;           // 0 or 1

    v8f acc[4][2];
    #pragma unroll
    for (int i = 0; i < 4; ++i)
        #pragma unroll
        for (int j = 0; j < 2; ++j)
            acc[i][j] = (v8f)(0.0f);

    // staging geometry: tile = 128 rows x 32 halves = 512 x b128 transfers;
    // each thread owns 2 transfers per matrix (4 async loads / tile).
    const int u0 = tid, u1 = tid + 256;
    const int r0 = u0 >> 2, c0 = (u0 & 3) << 3;
    const int r1 = u1 >> 2, c1 = (u1 & 3) << 3;

    const _Float16* gA[2] = { A + (size_t)(m0 + r0) * K + c0,
                              A + (size_t)(m0 + r1) * K + c1 };
    const _Float16* gB[2] = { B + (size_t)(n0 + r0) * K + c0,
                              B + (size_t)(n0 + r1) * K + c1 };
    unsigned lA[3][2], lB[3][2];
    #pragma unroll
    for (int b = 0; b < 3; ++b) {
        lA[b][0] = lds_off(&Ash[b][r0 * LDT + c0]);
        lA[b][1] = lds_off(&Ash[b][r1 * LDT + c1]);
        lB[b][0] = lds_off(&Bsh[b][r0 * LDT + c0]);
        lB[b][1] = lds_off(&Bsh[b][r1 * LDT + c1]);
    }

    auto stage = [&](int buf, int kt) {
        const int koff = kt * GK;
        #pragma unroll
        for (int i = 0; i < 2; ++i) {
            async_ld_b128(lA[buf][i], gA[i] + koff);
            async_ld_b128(lB[buf][i], gB[i] + koff);
        }
    };

    auto compute = [&](int buf) {
        v16h af[4], bf[2];
        // A layout (16-bit 16x32): lane m=l&15; halves 0..7 -> K = 8*lhi+0..7,
        // halves 8..15 -> K = 16 + 8*lhi + 0..7
        const int kloA = (lhi << 3);
        #pragma unroll
        for (int mt = 0; mt < 4; ++mt) {
            const _Float16* p = &Ash[buf][(wm + mt * 16 + l15) * LDT + kloA];
            uint4 x0 = *(const uint4*)p;
            uint4 x1 = *(const uint4*)(p + 16);
            union { v8u u; v16h h; } f;
            f.u = (v8u){x0.x, x0.y, x0.z, x0.w, x1.x, x1.y, x1.z, x1.w};
            af[mt] = f.h;
        }
        // B layout (16-bit 32x16): lane n=l&15; halves 0..7 -> K = 8*lhi+0..7,
        // halves 8..15 -> K = 16 + 8*lhi + 0..7
        const int kloB = (lhi << 3);
        #pragma unroll
        for (int nt = 0; nt < 2; ++nt) {
            const _Float16* p = &Bsh[buf][(wn + nt * 16 + l15) * LDT + kloB];
            uint4 x0 = *(const uint4*)p;
            uint4 x1 = *(const uint4*)(p + 16);
            union { v8u u; v16h h; } f;
            f.u = (v8u){x0.x, x0.y, x0.z, x0.w, x1.x, x1.y, x1.z, x1.w};
            bf[nt] = f.h;
        }
        #pragma unroll
        for (int mt = 0; mt < 4; ++mt)
            #pragma unroll
            for (int nt = 0; nt < 2; ++nt)
                acc[mt][nt] = __builtin_amdgcn_wmma_f32_16x16x32_f16(
                    false, af[mt], false, bf[nt],
                    (short)0, acc[mt][nt], false, false);
    };

    // prologue: stage tiles 0 and 1; wait until tile 0 is resident (<=4
    // outstanding leaves only tile 1 in flight).
    stage(0, 0);
    stage(1, 1);
    wait_asynccnt(1);
    __syncthreads();

    const int KT = K / GK;                 // 128 k-tiles
    int cur = 0;
    for (int kt = 0; kt < KT; ++kt) {
        const bool have2 = (kt + 2) < KT;
        if (have2) {
            int nb = cur + 2; if (nb >= 3) nb -= 3;
            stage(nb, kt + 2);             // fill buffer freed at kt-1
        }
        compute(cur);
        if (kt + 1 < KT) {
            // guarantee tile kt+1 is resident: in-order retirement means
            // <=4 outstanding == only the just-issued tile remains.
            wait_asynccnt(have2 ? 1 : 0);
            __syncthreads();
        }
        ++cur; if (cur >= 3) cur = 0;
    }

    // epilogue: C 16x16 f32 layout — VGPR v: lanes 0-15 M=v, lanes 16-31 M=v+8
    #pragma unroll
    for (int mt = 0; mt < 4; ++mt) {
        #pragma unroll
        for (int nt = 0; nt < 2; ++nt) {
            int n = n0 + wn + nt * 16 + l15;
            int mb = m0 + wm + mt * 16 + (lhi << 3);
            #pragma unroll
            for (int v = 0; v < 8; ++v)
                C[(size_t)(mb + v) * N + n] = acc[mt][nt][v];
        }
    }
}

// ---------------------------------------------------------------------------
extern "C" void kernel_launch(void* const* d_in, const int* in_sizes, int n_in,
                              void* d_out, int out_size, void* d_ws, size_t ws_size,
                              hipStream_t stream) {
    const float* x       = (const float*)d_in[0];
    const int*   trellis = (const int*)d_in[1];
    const float* tlut    = (const float*)d_in[2];
    const float* SU      = (const float*)d_in[3];
    const float* SV      = (const float*)d_in[4];
    float* out = (float*)d_out;

    char* ws = (char*)d_ws;
    _Float16* Xh  = (_Float16*)(ws);                          // 64 MiB
    _Float16* Wh  = (_Float16*)(ws + ((size_t)64 << 20));     // 32 MiB
    float*    w0  = (float*)   (ws + ((size_t)96 << 20));     // 64 MiB
    float*    w1  = (float*)   (ws + ((size_t)160 << 20));    // 64 MiB

    // 1) decode trellis -> w_hat (fp32)
    decode_kernel<<<NSEQ, 256, 0, stream>>>(trellis, tlut, w0);

    // 2) x fp32 -> f16
    cvt_kernel<<<(MROWS * IN_DIM) / (256 * 4), 256, 0, stream>>>(x, Xh);

    // 3) row FWHT (+SU) : w0 -> w1
    fwht_row_kernel<<<OUT_DIM, 256, 0, stream>>>(w0, w1, SU);

    // 4) col FWHT (+SV) + f16 : w1 -> Wh
    fwht_col_kernel<<<IN_DIM, 256, 0, stream>>>(w1, Wh, SV);

    // 5) y = Xh @ Wh^T  (WMMA f16 -> f32)
    dim3 grid(MROWS / GM, OUT_DIM / GN);
    gemm_kernel<<<grid, 256, 0, stream>>>(Xh, Wh, out);
}
